// PointerNetwork_54640573939932
// MI455X (gfx1250) — compile-verified, matrix operand order checked by reading, not run on hardware
//
#include <hip/hip_runtime.h>
#include <hip/hip_bf16.h>

#define Bn 128
#define Sn 256
#define Hn 256
#define BH 32768          // Bn*Hn
#define NW 262144         // 1024*256 (one 4H x H weight)

typedef _Float16 v16h __attribute__((ext_vector_type(16)));
typedef _Float16 v8h  __attribute__((ext_vector_type(8)));
typedef float    v8f  __attribute__((ext_vector_type(8)));
typedef unsigned int u32x4 __attribute__((ext_vector_type(4)));
typedef int          i32x8 __attribute__((ext_vector_type(8)));
typedef int          i32x4 __attribute__((ext_vector_type(4)));

struct P {
  // raw fp32 inputs
  const float* x; const int* y;
  const float *eWih0, *eb0, *eb1, *db0, *db1, *aV;
  const float *eWhh0, *eWih1, *eWhh1, *dWih0, *dWhh0, *dWih1, *dWhh1, *aW1, *aW2;
  // f16 weight copies (workspace)
  _Float16 *eWhh0h, *eWih1h, *eWhh1h, *dWih0h, *dWhh0h, *dWih1h, *dWhh1h, *aW1h, *aW2h;
  float* dWih0c;           // last column (dec_in) of dWih0, fp32
  // state / activations (workspace)
  _Float16 *encOut;        // [S][B][H] f16
  _Float16 *h0b, *h1b;     // [2][B][H] f16 double-buffered hidden states
  _Float16 *dif16;         // [B][H] context vector f16
  float *W1e;              // [S][B][32] f32
  float *c0, *c1;          // [B][H] f32 cell states
  float *q;                // [B][32]
  float *uj, *aj;          // [B][S]
  float *dec_in;           // [B]
  float *lossAcc;
  unsigned *encCtr, *decCtr;
  float* outBuf; int out_n;
};

// ---------------------------------------------------------------- helpers

__device__ __forceinline__ float sigm(float v) { return 1.0f / (1.0f + __expf(-v)); }

// Grid-wide sense-free barrier: monotonically increasing counter, target
// advances by gridDim.x per sync.  Counter is zeroed by k_init each launch.
__device__ __forceinline__ void grid_sync(unsigned* ctr, unsigned target) {
  __syncthreads();
  if (threadIdx.x == 0) {
    __threadfence();
    atomicAdd(ctr, 1u);
    while (__hip_atomic_load(ctr, __ATOMIC_RELAXED, __HIP_MEMORY_SCOPE_AGENT) < target) {
      __builtin_amdgcn_s_sleep(1);
    }
    __threadfence();
  }
  __syncthreads();
}

// Load one 16x32 f16 operand tile in the CDNA5 WMMA A-layout (ISA 7.12.2):
// lanes 0-15 row m: v[0..3]=K 8*kh+0..7, v[4..7]=K 16+8*kh+0..7 (kh = lane>>4).
// B operand is supplied as row-major N x K (i.e. B-transposed == PyTorch W).
__device__ __forceinline__ v16h ldA(const _Float16* base, int ld, int kt) {
  const int lane = threadIdx.x & 31;
  const int m = lane & 15, kh = lane >> 4;
  const _Float16* pp = base + (size_t)m * ld + kt * 32 + kh * 8;
  v8h lo = *(const v8h*)pp;
  v8h hi = *(const v8h*)(pp + 16);
  return __builtin_shufflevector(lo, hi, 0,1,2,3,4,5,6,7,8,9,10,11,12,13,14,15);
}

__device__ __forceinline__ v8f wmma16(v16h a, v16h b, v8f c) {
  return __builtin_amdgcn_wmma_f32_16x16x32_f16(false, a, false, b, (short)0, c,
                                                false, false);
}

// acc[g] += A[b0.., 0:256] * W[g*256+n0 .., 0:256]^T  (W in global, row-major NxK)
__device__ __forceinline__ void gemm4(v8f acc[4], const _Float16* A, int lda,
                                      const _Float16* W, int ldw, int n0) {
  for (int kt = 0; kt < 8; ++kt) {
    v16h a = ldA(A, lda, kt);
#pragma unroll
    for (int g = 0; g < 4; ++g) {
      v16h b = ldA(W + (size_t)(g * Hn + n0) * ldw, ldw, kt);
      acc[g] = wmma16(a, b, acc[g]);
    }
  }
}

// LDS variant: Wl is a [4][16][256] f16 slice staged into shared memory.
__device__ __forceinline__ void gemm4_lds(v8f acc[4], const _Float16* A, int lda,
                                          const _Float16* Wl) {
  for (int kt = 0; kt < 8; ++kt) {
    v16h a = ldA(A, lda, kt);
#pragma unroll
    for (int g = 0; g < 4; ++g) {
      v16h b = ldA(Wl + (size_t)g * 16 * Hn, Hn, kt);
      acc[g] = wmma16(a, b, acc[g]);
    }
  }
}

// ------------------------------------------------------- Tensor Data Mover
// Stage the 4 gate-tiles (16 rows each, K=256) of one [1024][256] f16 weight
// into LDS with a single TENSOR_LOAD_TO_LDS.  The tensor is viewed as
// [gate=4][row=256][k=256]; tile = (x=256, y=16, z=4), dim0_stride=256,
// dim1_stride=65536 (gate pitch).  D# bitfields per CDNA5 ISA.
// This toolchain's builtin is the 6-arg form (clang-23 / therock-10.0):
//   (u32x4 g0, i32x8 g1, i32x4 g2, i32x4 g3, i32x8 g4, i32 cpol)
#if __has_builtin(__builtin_amdgcn_tensor_load_to_lds)
#define HAVE_TDM 1
__device__ __forceinline__ void tdm_load_w(unsigned ldsOff, const _Float16* gptr) {
  const unsigned long long ga = (unsigned long long)(uintptr_t)gptr;
  u32x4 g0;
  g0[0] = 1u;                                   // count=1 (valid user D#)
  g0[1] = ldsOff;                               // lds_addr (bytes)
  g0[2] = (unsigned)(ga & 0xffffffffu);         // global_addr[31:0]
  g0[3] = (unsigned)((ga >> 32) & 0x1ffffffu)   // global_addr[56:32]
        | (2u << 30);                           // type = 2 ("image")
  i32x8 g1;
  g1[0] = (int)(1u << 16);                      // data_size = 1 (2 bytes)
  g1[1] = (int)(256u << 16);                    // tensor_dim0[15:0] = 256
  g1[2] = (int)(256u << 16);                    // tensor_dim1[15:0] = 256
  g1[3] = (int)(256u << 16);                    // tile_dim0 = 256
  g1[4] = (int)(16u | (4u << 16));              // tile_dim1 = 16, tile_dim2 = 4
  g1[5] = (int)256;                             // tensor_dim0_stride = 256
  g1[6] = 0;                                    // stride0 hi | stride1 lo16 (65536&0xffff=0)
  g1[7] = (int)(65536u >> 16);                  // tensor_dim1_stride hi = 1
  i32x4 g2;
  g2[0] = 4;                                    // tensor_dim2 = 4 (gates)
  g2[1] = 0; g2[2] = 0; g2[3] = 0;
  i32x4 g3; g3[0] = 0; g3[1] = 0; g3[2] = 0; g3[3] = 0;
  i32x8 g4; g4[0] = 0; g4[1] = 0; g4[2] = 0; g4[3] = 0;
  g4[4] = 0; g4[5] = 0; g4[6] = 0; g4[7] = 0;   // aux group (zero-filled)
  __builtin_amdgcn_tensor_load_to_lds(g0, g1, g2, g3, g4, 0);
}
#else
#define HAVE_TDM 0
#endif

// ---------------------------------------------------------------- init

__global__ void __launch_bounds__(256) k_init(P p) {
  const size_t i0 = blockIdx.x * (size_t)blockDim.x + threadIdx.x;
  const size_t st = (size_t)gridDim.x * blockDim.x;
  for (size_t k = i0; k < NW; k += st) {
    p.eWhh0h[k] = (_Float16)p.eWhh0[k];
    p.eWih1h[k] = (_Float16)p.eWih1[k];
    p.eWhh1h[k] = (_Float16)p.eWhh1[k];
    p.dWhh0h[k] = (_Float16)p.dWhh0[k];
    p.dWih1h[k] = (_Float16)p.dWih1[k];
    p.dWhh1h[k] = (_Float16)p.dWhh1[k];
    const size_t r = k >> 8, c = k & 255;          // dWih0 is [1024][257]
    p.dWih0h[r * 264 + c] = (_Float16)p.dWih0[r * 257 + c];  // pad ld=264 (16B aligned)
  }
  for (size_t k = i0; k < 4 * Hn; k += st) p.dWih0c[k] = p.dWih0[k * 257 + 256];
  for (size_t k = i0; k < 32 * Hn; k += st) {      // pad 20 -> 32 rows with zeros
    const size_t r = k >> 8;
    p.aW1h[k] = (r < 20) ? (_Float16)p.aW1[k] : (_Float16)0.0f;
    p.aW2h[k] = (r < 20) ? (_Float16)p.aW2[k] : (_Float16)0.0f;
  }
  for (size_t k = i0; k < 2 * BH; k += st) {
    p.h0b[k] = (_Float16)0.0f; p.h1b[k] = (_Float16)0.0f;
  }
  for (size_t k = i0; k < BH; k += st) { p.c0[k] = 0.0f; p.c1[k] = 0.0f; }
  for (size_t k = i0; k < Bn; k += st) p.dec_in[k] = 0.0f;
  if (i0 == 0) { *p.lossAcc = 0.0f; *p.encCtr = 0u; *p.decCtr = 0u; }
}

// ---------------------------------------------------------------- encoder
// 16 WGs x 8 waves.  WG = one 16-wide hidden-column tile, wave = one 16-row
// batch tile.  Recurrent weights for this WG's columns (96KB) are staged into
// LDS once by the Tensor Data Mover and reused for all 256 timesteps.

__global__ void __launch_bounds__(256) k_encoder(P p) {
  const int wg = blockIdx.x;                  // 0..15 -> n0
  const int wave = threadIdx.x >> 5;          // 0..7  -> b0
  const int lane = threadIdx.x & 31;
  const int n0 = wg * 16, b0 = wave * 16;
  const int col = lane & 15, kh = lane >> 4;
  const int n = n0 + col;
  unsigned tgt = 0;

  __shared__ _Float16 smW[3 * 4 * 16 * Hn];   // 96KB: [3 weights][4 gates][16][256]

#if HAVE_TDM
  if (wave == 0) {                            // one wave issues the DMA
    const unsigned ldsBase = (unsigned)(uintptr_t)(void*)smW;
    tdm_load_w(ldsBase +     0u, p.eWhh0h + (size_t)n0 * Hn);
    tdm_load_w(ldsBase + 32768u, p.eWih1h + (size_t)n0 * Hn);
    tdm_load_w(ldsBase + 65536u, p.eWhh1h + (size_t)n0 * Hn);
    __builtin_amdgcn_s_wait_tensorcnt(0);
  }
#else
  {
    const _Float16* src[3] = { p.eWhh0h, p.eWih1h, p.eWhh1h };
    for (int sl = 0; sl < 3; ++sl)
      for (int i = threadIdx.x; i < 4 * 16 * Hn; i += 256) {
        const int g = i >> 12, rm = (i >> 8) & 15, k = i & 255;
        smW[sl * 16384 + i] = src[sl][(size_t)(g * Hn + n0 + rm) * Hn + k];
      }
  }
#endif
  __syncthreads();

  float wx[4], bv0[4], bv1[4];
#pragma unroll
  for (int g = 0; g < 4; ++g) {
    wx[g]  = p.eWih0[g * Hn + n];             // [4H,1] input weight column
    bv0[g] = p.eb0[g * Hn + n];
    bv1[g] = p.eb1[g * Hn + n];
  }

  for (int t = 0; t < Sn; ++t) {
    const _Float16* h0cur = p.h0b + (t & 1) * BH;
    _Float16*       h0nxt = p.h0b + ((t + 1) & 1) * BH;
    const _Float16* h1cur = p.h1b + (t & 1) * BH;
    _Float16*       h1nxt = p.h1b + ((t + 1) & 1) * BH;

    { // layer 0: gates = x_t * Wih0 + h0 @ Whh0^T + b
      v8f acc[4] = {};
      gemm4_lds(acc, h0cur + (size_t)b0 * Hn, Hn, smW);
#pragma unroll
      for (int r = 0; r < 8; ++r) {
        const int b = b0 + kh * 8 + r;
        const float xt = p.x[b * Sn + t];
        const float iv = acc[0][r] + bv0[0] + xt * wx[0];
        const float fv = acc[1][r] + bv0[1] + xt * wx[1];
        const float gv = acc[2][r] + bv0[2] + xt * wx[2];
        const float ov = acc[3][r] + bv0[3] + xt * wx[3];
        const float cn = sigm(fv) * p.c0[b * Hn + n] + sigm(iv) * tanhf(gv);
        const float hn = sigm(ov) * tanhf(cn);
        p.c0[b * Hn + n] = cn;
        h0nxt[b * Hn + n] = (_Float16)hn;
      }
    }
    grid_sync(p.encCtr, tgt += gridDim.x);

    { // layer 1: gates = h0_new @ Wih1^T + h1 @ Whh1^T + b
      v8f acc[4] = {};
      gemm4_lds(acc, h0nxt + (size_t)b0 * Hn, Hn, smW + 16384);
      gemm4_lds(acc, h1cur + (size_t)b0 * Hn, Hn, smW + 32768);
#pragma unroll
      for (int r = 0; r < 8; ++r) {
        const int b = b0 + kh * 8 + r;
        const float iv = acc[0][r] + bv1[0];
        const float fv = acc[1][r] + bv1[1];
        const float gv = acc[2][r] + bv1[2];
        const float ov = acc[3][r] + bv1[3];
        const float cn = sigm(fv) * p.c1[b * Hn + n] + sigm(iv) * tanhf(gv);
        const float hn = sigm(ov) * tanhf(cn);
        p.c1[b * Hn + n] = cn;
        const _Float16 hh = (_Float16)hn;
        h1nxt[b * Hn + n] = hh;
        p.encOut[(size_t)t * BH + b * Hn + n] = hh;
      }
    }
    grid_sync(p.encCtr, tgt += gridDim.x);
  }
}

// ---------------------------------------------------------------- W1e = enc_out @ aW1^T  ([S][B][32])

__global__ void __launch_bounds__(256) k_w1e(P p) {
  const int s = blockIdx.x;
  const int wave = threadIdx.x >> 5, lane = threadIdx.x & 31;
  const int b0 = wave * 16, col = lane & 15, kh = lane >> 4;
  const _Float16* A = p.encOut + (size_t)s * BH + (size_t)b0 * Hn;
  float* o = p.W1e + (size_t)s * Bn * 32;
#pragma unroll
  for (int nt = 0; nt < 2; ++nt) {
    v8f acc = {};
    for (int kt = 0; kt < 8; ++kt) {
      v16h a = ldA(A, Hn, kt);
      v16h b = ldA(p.aW1h + (size_t)nt * 16 * Hn, Hn, kt);
      acc = wmma16(a, b, acc);
    }
#pragma unroll
    for (int r = 0; r < 8; ++r)
      o[(b0 + kh * 8 + r) * 32 + nt * 16 + col] = acc[r];
  }
}

// ---------------------------------------------------------------- decoder
// 32 WGs x 8 waves, persistent over 256 steps with 6 grid syncs per step.
// Decoder weights stay in (192MB) L2: each is re-read once/step, shared by all waves.

__global__ void __launch_bounds__(256) k_decoder(P p) {
  const int gthread  = blockIdx.x * blockDim.x + threadIdx.x;
  const int nthreads = gridDim.x * blockDim.x;
  const int waveId = gthread >> 5;
  const int lane = threadIdx.x & 31;
  const int col = lane & 15, kh = lane >> 4;
  unsigned tgt = 0;

  // L2 warm hints for decoder weights (global_prefetch_b8)
  for (size_t o = (size_t)gthread * 512; o < (size_t)NW; o += (size_t)nthreads * 512) {
    __builtin_prefetch(p.dWhh0h + o, 0, 1);
    __builtin_prefetch(p.dWih1h + o, 0, 1);
  }

  for (int t = 0; t < Sn; ++t) {
    const _Float16* h0cur = p.h0b + (t & 1) * BH;     // encoder finals feed t=0
    _Float16*       h0nxt = p.h0b + ((t + 1) & 1) * BH;
    const _Float16* h1cur = p.h1b + (t & 1) * BH;
    _Float16*       h1nxt = p.h1b + ((t + 1) & 1) * BH;

    // A: q = h[0] @ aW2^T  (8 M-tiles x 2 N-tiles = waves 0..15)
    if (waveId < 16) {
      const int mt = waveId >> 1, nt = waveId & 1;
      v8f acc = {};
      for (int kt = 0; kt < 8; ++kt) {
        v16h a = ldA(h0cur + (size_t)mt * 16 * Hn, Hn, kt);
        v16h b = ldA(p.aW2h + (size_t)nt * 16 * Hn, Hn, kt);
        acc = wmma16(a, b, acc);
      }
#pragma unroll
      for (int r = 0; r < 8; ++r)
        p.q[(mt * 16 + kh * 8 + r) * 32 + nt * 16 + col] = acc[r];
    }
    grid_sync(p.decCtr, tgt += gridDim.x);

    // B: uj[b][s] = sum_u v[u] * tanh(W1e[s][b][u] + q[b][u])
    for (int i = gthread; i < Bn * Sn; i += nthreads) {
      const int b = i >> 8, s = i & 255;
      const float* w1 = p.W1e + ((size_t)s * Bn + b) * 32;
      const float* qq = p.q + b * 32;
      float a2 = 0.0f;
#pragma unroll
      for (int u = 0; u < 20; ++u) a2 += p.aV[u] * tanhf(w1[u] + qq[u]);
      p.uj[b * Sn + s] = a2;
    }
    grid_sync(p.decCtr, tgt += gridDim.x);

    // C: per-row softmax / argmax / CE loss (one wave per batch row)
    if (waveId < Bn) {
      const int b = waveId;
      float uu[8]; float vmax = -1e30f; int amax = 0;
#pragma unroll
      for (int j = 0; j < 8; ++j) {
        const int s = j * 32 + lane;
        uu[j] = p.uj[b * Sn + s];
        if (uu[j] > vmax) { vmax = uu[j]; amax = s; }   // keeps first max
      }
#pragma unroll
      for (int off = 16; off >= 1; off >>= 1) {
        const float ov = __shfl_xor(vmax, off, 32);
        const int   oi = __shfl_xor(amax, off, 32);
        if (ov > vmax || (ov == vmax && oi < amax)) { vmax = ov; amax = oi; }
      }
      float ssum = 0.0f;
#pragma unroll
      for (int j = 0; j < 8; ++j) ssum += __expf(uu[j] - vmax);
#pragma unroll
      for (int off = 16; off >= 1; off >>= 1) ssum += __shfl_xor(ssum, off, 32);
      const float inv = 1.0f / ssum;
#pragma unroll
      for (int j = 0; j < 8; ++j)
        p.aj[b * Sn + j * 32 + lane] = __expf(uu[j] - vmax) * inv;
      if (lane == 0) {
        const float logZ = __logf(ssum) + vmax;
        const int yv = p.y[b * Sn + t];
        atomicAdd(p.lossAcc, (logZ - p.uj[b * Sn + yv]) * (1.0f / 16384.0f));
        if (t * Bn + b < p.out_n) p.outBuf[t * Bn + b] = (float)amax;
      }
    }
    grid_sync(p.decCtr, tgt += gridDim.x);

    // D: context di[b][h] = sum_s aj[b][s] * enc_out[s][b][h]
    for (int i = gthread; i < BH; i += nthreads) {
      const int b = i >> 8, h = i & 255;
      const float* ajr = p.aj + b * Sn;
      const _Float16* eo = p.encOut + (size_t)b * Hn + h;
      float a3 = 0.0f;
      for (int s = 0; s < Sn; ++s) a3 += ajr[s] * (float)eo[(size_t)s * BH];
      p.dif16[b * Hn + h] = (_Float16)a3;
    }
    grid_sync(p.decCtr, tgt += gridDim.x);

    // E0: decoder LSTM layer 0 (input [di, dec_in], K=256 WMMA + rank-1 tail)
    if (waveId < 128) {
      const int mt = waveId >> 4, nt2 = waveId & 15;
      const int b0 = mt * 16, n0 = nt2 * 16, n = n0 + col;
      v8f acc[4] = {};
      gemm4(acc, p.dif16 + (size_t)b0 * Hn, Hn, p.dWih0h, 264, n0);
      gemm4(acc, h0cur + (size_t)b0 * Hn, Hn, p.dWhh0h, Hn, n0);
      float wc[4], bb[4];
#pragma unroll
      for (int g = 0; g < 4; ++g) {
        wc[g] = p.dWih0c[g * Hn + n]; bb[g] = p.db0[g * Hn + n];
      }
#pragma unroll
      for (int r = 0; r < 8; ++r) {
        const int b = b0 + kh * 8 + r;
        const float din = p.dec_in[b];
        const float iv = acc[0][r] + bb[0] + din * wc[0];
        const float fv = acc[1][r] + bb[1] + din * wc[1];
        const float gv = acc[2][r] + bb[2] + din * wc[2];
        const float ov = acc[3][r] + bb[3] + din * wc[3];
        const float cn = sigm(fv) * p.c0[b * Hn + n] + sigm(iv) * tanhf(gv);
        const float hn = sigm(ov) * tanhf(cn);
        p.c0[b * Hn + n] = cn;
        h0nxt[b * Hn + n] = (_Float16)hn;
      }
    }
    grid_sync(p.decCtr, tgt += gridDim.x);

    // E1: decoder LSTM layer 1
    if (waveId < 128) {
      const int mt = waveId >> 4, nt2 = waveId & 15;
      const int b0 = mt * 16, n0 = nt2 * 16, n = n0 + col;
      v8f acc[4] = {};
      gemm4(acc, h0nxt + (size_t)b0 * Hn, Hn, p.dWih1h, Hn, n0);
      gemm4(acc, h1cur + (size_t)b0 * Hn, Hn, p.dWhh1h, Hn, n0);
#pragma unroll
      for (int r = 0; r < 8; ++r) {
        const int b = b0 + kh * 8 + r;
        const float iv = acc[0][r] + p.db1[n];
        const float fv = acc[1][r] + p.db1[Hn + n];
        const float gv = acc[2][r] + p.db1[2 * Hn + n];
        const float ov = acc[3][r] + p.db1[3 * Hn + n];
        const float cn = sigm(fv) * p.c1[b * Hn + n] + sigm(iv) * tanhf(gv);
        const float hn = sigm(ov) * tanhf(cn);
        p.c1[b * Hn + n] = cn;
        h1nxt[b * Hn + n] = (_Float16)hn;
      }
    }
    // F: teacher forcing — next dec_in = x[b, y[b,t]]
    for (int i = gthread; i < Bn; i += nthreads) {
      const int yv = p.y[i * Sn + t];
      p.dec_in[i] = p.x[i * Sn + yv];
    }
    grid_sync(p.decCtr, tgt += gridDim.x);
  }

  if (blockIdx.x == 0 && threadIdx.x == 0 && p.out_n > Sn * Bn)
    p.outBuf[Sn * Bn] = *p.lossAcc;
}

// ---------------------------------------------------------------- host

extern "C" void kernel_launch(void* const* d_in, const int* in_sizes, int n_in,
                              void* d_out, int out_size, void* d_ws, size_t ws_size,
                              hipStream_t stream) {
  (void)in_sizes; (void)n_in; (void)ws_size;
  char* w = (char*)d_ws;
  size_t off = 0;
  auto alloc = [&](size_t bytes) -> void* {
    void* ptr = w + off;
    off = (off + bytes + 255) & ~(size_t)255;
    return ptr;
  };

  P p;
  p.x     = (const float*)d_in[0];  p.y     = (const int*)d_in[1];
  p.eWih0 = (const float*)d_in[2];  p.eWhh0 = (const float*)d_in[3];
  p.eb0   = (const float*)d_in[4];  p.eWih1 = (const float*)d_in[5];
  p.eWhh1 = (const float*)d_in[6];  p.eb1   = (const float*)d_in[7];
  p.dWih0 = (const float*)d_in[8];  p.dWhh0 = (const float*)d_in[9];
  p.db0   = (const float*)d_in[10]; p.dWih1 = (const float*)d_in[11];
  p.dWhh1 = (const float*)d_in[12]; p.db1   = (const float*)d_in[13];
  p.aW1   = (const float*)d_in[14]; p.aW2   = (const float*)d_in[15];
  p.aV    = (const float*)d_in[16];

  p.eWhh0h = (_Float16*)alloc((size_t)NW * 2);
  p.eWih1h = (_Float16*)alloc((size_t)NW * 2);
  p.eWhh1h = (_Float16*)alloc((size_t)NW * 2);
  p.dWih0h = (_Float16*)alloc((size_t)1024 * 264 * 2);
  p.dWhh0h = (_Float16*)alloc((size_t)NW * 2);
  p.dWih1h = (_Float16*)alloc((size_t)NW * 2);
  p.dWhh1h = (_Float16*)alloc((size_t)NW * 2);
  p.aW1h   = (_Float16*)alloc((size_t)32 * Hn * 2);
  p.aW2h   = (_Float16*)alloc((size_t)32 * Hn * 2);
  p.dWih0c = (float*)alloc((size_t)4 * Hn * 4);
  p.encOut = (_Float16*)alloc((size_t)Sn * BH * 2);
  p.W1e    = (float*)alloc((size_t)Sn * Bn * 32 * 4);
  p.h0b    = (_Float16*)alloc((size_t)2 * BH * 2);
  p.h1b    = (_Float16*)alloc((size_t)2 * BH * 2);
  p.dif16  = (_Float16*)alloc((size_t)BH * 2);
  p.c0     = (float*)alloc((size_t)BH * 4);
  p.c1     = (float*)alloc((size_t)BH * 4);
  p.q      = (float*)alloc((size_t)Bn * 32 * 4);
  p.uj     = (float*)alloc((size_t)Bn * Sn * 4);
  p.aj     = (float*)alloc((size_t)Bn * Sn * 4);
  p.dec_in = (float*)alloc((size_t)Bn * 4);
  p.lossAcc = (float*)alloc(256);
  p.encCtr  = (unsigned*)alloc(256);
  p.decCtr  = (unsigned*)alloc(256);
  p.outBuf = (float*)d_out;
  p.out_n  = out_size;

  k_init<<<256, 256, 0, stream>>>(p);
  k_encoder<<<16, 256, 0, stream>>>(p);   // persistent, TDM-staged weights
  k_w1e<<<Sn, 256, 0, stream>>>(p);
  k_decoder<<<32, 256, 0, stream>>>(p);   // persistent, 6 syncs/step
}
